// Dim6RotStructureHead_10325101379768
// MI455X (gfx1250) — compile-verified, matrix operand order checked by reading, not run on hardware
//
#include <hip/hip_runtime.h>
#include <hip/hip_bf16.h>
#include <math.h>

// Problem dims (compile-time constants from the reference)
#define BSZ   4
#define LSZ   1024
#define DSZ   1536                 // D == N == K
#define MSZ   (BSZ * LSZ)          // 4096 tokens
#define OUTSZ 23
#define AFFINE_OUT_ELEMS (MSZ * 12)

// GEMM tiling
#define TILE_M 128
#define TILE_N 64
#define KSTEP  32
#define NSTEPS (DSZ / KSTEP)       // 48
#define ROW_BYTES 80               // 32 bf16 = 64B data, padded to 80B (bank-conflict-free)
#define A_BYTES (TILE_M * ROW_BYTES)   // 10240
#define B_BYTES (TILE_N * ROW_BYTES)   // 5120
#define BUF_BYTES (A_BYTES + B_BYTES)  // 15360
#define SMEM_BYTES (2 * BUF_BYTES)     // 30720 (double buffered)

typedef __attribute__((ext_vector_type(16))) __bf16 v16bf;
typedef __attribute__((ext_vector_type(8)))  float  v8f;

union FragBF { uint4 u4[2]; v16bf v; };

__device__ __forceinline__ unsigned short f32_to_bf16_rne(float f) {
  union { float f; unsigned int u; } x; x.f = f;
  unsigned int r = x.u + 0x7FFFu + ((x.u >> 16) & 1u);
  return (unsigned short)(r >> 16);
}

// gfx1250 async global->LDS copy (ASYNCcnt-tracked DMA, no VGPR data path).
// First operand = LDS byte offset (low 32 bits of generic shared pointer),
// second = 64-bit global address.
__device__ __forceinline__ void async_g2l_b128(unsigned lds_off, const void* g) {
  asm volatile("global_load_async_to_lds_b128 %0, %1, off"
               :: "v"(lds_off), "v"(g) : "memory");
}

// ---------------- Kernel 0: f32 -> bf16 convert (RNE) ----------------
__global__ __launch_bounds__(256)
void convert_f32_to_bf16(const float* __restrict__ in,
                         unsigned short* __restrict__ out, int n) {
  int i = blockIdx.x * blockDim.x + threadIdx.x;
  int stride = gridDim.x * blockDim.x;
  for (; i < n; i += stride) out[i] = f32_to_bf16_rne(in[i]);
}

// ---------------- Kernel 1: h = gelu(x @ w1^T + b1) via WMMA bf16 ----------------
// Block: 256 threads = 8 wave32s. Tile 128(M) x 64(N), K-step 32, LDS double-buffer
// filled with global_load_async_to_lds_b128, consumed as ds_load_b128 fragments.
// Wave w: n-sub = w&3, m-subs = (w>>2)*4 + {0..3}  -> 4 accumulators per wave.
// K-loop kept rolled (unroll 1) with pointer-swapped buffers so accumulator phis
// stay pinned in one register set (no cross-body v_mov copies / hazard NOPs).
__global__ __launch_bounds__(256)
void gemm1_gelu(const __bf16* __restrict__ xb,    // MSZ x DSZ, row-major
                const __bf16* __restrict__ w1b,   // DSZ x DSZ, row-major (N x K)
                const float*  __restrict__ b1,
                float* __restrict__ h) {          // MSZ x DSZ
  __shared__ __align__(16) uint4 smem4[SMEM_BYTES / 16];
  char* smem = (char*)smem4;

  const int tid  = threadIdx.x;
  const int wave = tid >> 5;
  const int lane = tid & 31;
  const int lr   = lane & 15;
  const int half = lane >> 4;

  const int nsub   = wave & 3;
  const int mgroup = wave >> 2;                 // 0..1
  const int tile_m = blockIdx.y * TILE_M;
  const int tile_n = blockIdx.x * TILE_N;

  // ---- staging addresses (per thread: 2 A chunks + 1 B chunk of 16B) ----
  const int ca0_row = tid >> 2;                 // A chunk 0: rows 0..63
  const int ca1_row = (tid + 256) >> 2;         // A chunk 1: rows 64..127
  const int ca_sub  = tid & 3;                  // 16B sub-chunk within 64B row
  const int cb_row  = tid >> 2;                 // B rows 0..63
  const __bf16* gA0 = xb  + (size_t)(tile_m + ca0_row) * DSZ + ca_sub * 8;
  const __bf16* gA1 = xb  + (size_t)(tile_m + ca1_row) * DSZ + ca_sub * 8;
  const __bf16* gB  = w1b + (size_t)(tile_n + cb_row)  * DSZ + ca_sub * 8;

  // per-thread LDS byte offsets (relative to buffer base)
  const unsigned stA0 = (unsigned)(ca0_row * ROW_BYTES + ca_sub * 16);
  const unsigned stA1 = (unsigned)(ca1_row * ROW_BYTES + ca_sub * 16);
  const unsigned stB  = (unsigned)(A_BYTES + cb_row * ROW_BYTES + ca_sub * 16);
  // per-lane consume offsets (relative to buffer base)
  const unsigned ldB  = (unsigned)(A_BYTES + (nsub * 16 + lr) * ROW_BYTES + half * 16);
  const unsigned ldA  = (unsigned)((mgroup * 64 + lr) * ROW_BYTES + half * 16);

  v8f acc[4];
#pragma unroll
  for (int i = 0; i < 4; ++i) acc[i] = (v8f){};

  char* cur = smem;               // buffer being consumed this iteration
  char* nxt = smem + BUF_BYTES;   // buffer being staged this iteration

  // prologue: stage K-slab 0 into `cur`
  async_g2l_b128((unsigned)(size_t)(cur) + stA0, gA0);
  async_g2l_b128((unsigned)(size_t)(cur) + stA1, gA1);
  async_g2l_b128((unsigned)(size_t)(cur) + stB,  gB);

#pragma unroll 1
  for (int ks = 0; ks < NSTEPS; ++ks) {
    if (ks + 1 < NSTEPS) {
      // stage next K-slab into `nxt`, then wait for current slab's 3 ops
      const int k1 = (ks + 1) * KSTEP;
      async_g2l_b128((unsigned)(size_t)(nxt) + stA0, gA0 + k1);
      async_g2l_b128((unsigned)(size_t)(nxt) + stA1, gA1 + k1);
      async_g2l_b128((unsigned)(size_t)(nxt) + stB,  gB  + k1);
      asm volatile("s_wait_asynccnt 0x3" ::: "memory");
    } else {
      asm volatile("s_wait_asynccnt 0x0" ::: "memory");
    }
    __syncthreads();

    // ---- consume current buffer ----
    // B fragment (shared by the wave's 4 accumulators)
    FragBF bf;
    const char* bp = cur + ldB;
    bf.u4[0] = *(const uint4*)bp;
    bf.u4[1] = *(const uint4*)(bp + 32);

#pragma unroll
    for (int i = 0; i < 4; ++i) {
      FragBF af;
      const char* ap = cur + ldA + i * (16 * ROW_BYTES);
      af.u4[0] = *(const uint4*)ap;
      af.u4[1] = *(const uint4*)(ap + 32);
      acc[i] = __builtin_amdgcn_wmma_f32_16x16x32_bf16(
          false, af.v, false, bf.v, (short)0, acc[i], false, false);
    }
    __syncthreads();

    // swap buffers
    char* tmp = cur; cur = nxt; nxt = tmp;
  }

  // Epilogue: + bias, exact GELU, store f32.
  // C/D layout: VGPR r, lane<16 -> (M=r, N=lane), lane>=16 -> (M=r+8, N=lane-16)
  const int ocol = tile_n + nsub * 16 + lr;
  const float bias = b1[ocol];
  const float inv_sqrt2 = 0.70710678118654752f;
#pragma unroll
  for (int i = 0; i < 4; ++i) {
    const int mbase = tile_m + (mgroup * 4 + i) * 16;
#pragma unroll
    for (int r = 0; r < 8; ++r) {
      const int orow = mbase + r + 8 * half;
      float v = acc[i][r] + bias;
      h[(size_t)orow * DSZ + ocol] = 0.5f * v * (1.0f + erff(v * inv_sqrt2));
    }
  }
}

// ---------------- Kernel 2: LayerNorm + small projection + rotation head ----------------
__global__ __launch_bounds__(256)
void head_kernel(const float* __restrict__ h,
                 const float* __restrict__ ln_g,
                 const float* __restrict__ ln_b,
                 const float* __restrict__ w2,   // OUTSZ x DSZ
                 const float* __restrict__ b2,
                 const float* __restrict__ affine,      // MSZ x 12
                 const unsigned char* __restrict__ amask,
                 float* __restrict__ out) {
  __shared__ float hs[DSZ];
  __shared__ float redA[8], redB[8];
  __shared__ float pv[24];
  __shared__ float stats[2];

  const int t    = blockIdx.x;     // token index in [0, MSZ)
  const int tid  = threadIdx.x;
  const int lane = tid & 31;
  const int wave = tid >> 5;

  // Pass 1: load row, accumulate sum / sumsq
  float s = 0.f, ss = 0.f;
  for (int i = tid; i < DSZ; i += 256) {
    float v = h[(size_t)t * DSZ + i];
    hs[i] = v;
    s += v; ss += v * v;
  }
  for (int off = 16; off > 0; off >>= 1) {
    s  += __shfl_down(s,  off, 32);
    ss += __shfl_down(ss, off, 32);
  }
  if (lane == 0) { redA[wave] = s; redB[wave] = ss; }
  __syncthreads();
  if (tid == 0) {
    float S = 0.f, SS = 0.f;
    for (int i = 0; i < 8; ++i) { S += redA[i]; SS += redB[i]; }
    float mu  = S / (float)DSZ;
    float var = SS / (float)DSZ - mu * mu;
    stats[0] = mu;
    stats[1] = rsqrtf(var + 1e-5f);
  }
  __syncthreads();

  const float mu = stats[0], rstd = stats[1];
  for (int i = tid; i < DSZ; i += 256)
    hs[i] = (hs[i] - mu) * rstd * ln_g[i] + ln_b[i];
  __syncthreads();

  // p[o] = hs . w2[o,:] + b2[o]; wave w owns outputs 3w..3w+2
  for (int o = wave * 3; o < wave * 3 + 3 && o < OUTSZ; ++o) {
    float acc = 0.f;
    const float* wr = w2 + (size_t)o * DSZ;
    for (int i = lane; i < DSZ; i += 32) acc += hs[i] * wr[i];
    for (int off = 16; off > 0; off >>= 1) acc += __shfl_down(acc, off, 32);
    if (lane == 0) pv[o] = acc + b2[o];
  }
  __syncthreads();

  if (tid == 0) {
    float tr0 = pv[0] * 10.f, tr1 = pv[1] * 10.f, tr2 = pv[2] * 10.f;
    float xv0 = pv[3], xv1 = pv[4], xv2 = pv[5];
    float yv0 = pv[6], yv1 = pv[7], yv2 = pv[8];
    float nx = sqrtf(xv0*xv0 + xv1*xv1 + xv2*xv2) + 1e-5f;
    xv0 /= nx; xv1 /= nx; xv2 /= nx;
    float ny = sqrtf(yv0*yv0 + yv1*yv1 + yv2*yv2) + 1e-5f;
    yv0 /= ny; yv1 /= ny; yv2 /= ny;

    // Gram-Schmidt on (trans - neg_x, p_xy - trans) == (-xv, yv)
    float g0 = sqrtf(xv0*xv0 + xv1*xv1 + xv2*xv2 + 1e-12f);
    float e0x = -xv0 / g0, e0y = -xv1 / g0, e0z = -xv2 / g0;
    float dd = e0x*yv0 + e0y*yv1 + e0z*yv2;
    float f0 = yv0 - e0x*dd, f1 = yv1 - e0y*dd, f2 = yv2 - e0z*dd;
    float g1 = sqrtf(f0*f0 + f1*f1 + f2*f2 + 1e-12f);
    f0 /= g1; f1 /= g1; f2 /= g1;
    float c0 = e0y*f2 - e0z*f1;
    float c1 = e0z*f0 - e0x*f2;
    float c2 = e0x*f1 - e0y*f0;

    // Ru columns are (e0, e1, e2)
    float Ru[3][3] = { { e0x, f0, c0 },
                       { e0y, f1, c1 },
                       { e0z, f2, c2 } };
    float tu[3] = { tr0, tr1, tr2 };

    if (!amask[t]) {
      Ru[0][0] = 1.f; Ru[0][1] = 0.f; Ru[0][2] = 0.f;
      Ru[1][0] = 0.f; Ru[1][1] = 1.f; Ru[1][2] = 0.f;
      Ru[2][0] = 0.f; Ru[2][1] = 0.f; Ru[2][2] = 1.f;
      tu[0] = tu[1] = tu[2] = 0.f;
    }

    const float* A = affine + (size_t)t * 12;
    float R[3][3], tt[3];
#pragma unroll
    for (int i = 0; i < 3; ++i) {
#pragma unroll
      for (int j = 0; j < 3; ++j)
        R[i][j] = A[i*3+0]*Ru[0][j] + A[i*3+1]*Ru[1][j] + A[i*3+2]*Ru[2][j];
      tt[i] = A[i*3+0]*tu[0] + A[i*3+1]*tu[1] + A[i*3+2]*tu[2] + A[9+i];
    }

    float* oa = out + (size_t)t * 12;
#pragma unroll
    for (int i = 0; i < 3; ++i) {
#pragma unroll
      for (int j = 0; j < 3; ++j) oa[i*3+j] = R[i][j];
      oa[9+i] = tt[i];
    }

    const float bb[3][3] = { { -0.525f, 1.363f, 0.f },
                             {  0.f,    0.f,    0.f },
                             {  1.526f, 0.f,    0.f } };
    float* ox = out + AFFINE_OUT_ELEMS + (size_t)t * 9;
#pragma unroll
    for (int a = 0; a < 3; ++a)
#pragma unroll
      for (int i = 0; i < 3; ++i)
        ox[a*3+i] = R[i][0]*bb[a][0] + R[i][1]*bb[a][1] + R[i][2]*bb[a][2] + tt[i];
  }
}

// ---------------- Host launcher ----------------
extern "C" void kernel_launch(void* const* d_in, const int* in_sizes, int n_in,
                              void* d_out, int out_size, void* d_ws, size_t ws_size,
                              hipStream_t stream) {
  const float*         x      = (const float*)d_in[0];
  const float*         affine = (const float*)d_in[1];
  const unsigned char* amask  = (const unsigned char*)d_in[2];
  const float*         w1     = (const float*)d_in[3];
  const float*         b1     = (const float*)d_in[4];
  const float*         ln_g   = (const float*)d_in[5];
  const float*         ln_b   = (const float*)d_in[6];
  const float*         w2     = (const float*)d_in[7];
  const float*         b2     = (const float*)d_in[8];
  float* out = (float*)d_out;

  char* ws = (char*)d_ws;
  unsigned short* xb  = (unsigned short*)ws;                                  // 12.6 MB
  unsigned short* w1b = (unsigned short*)(ws + (size_t)MSZ * DSZ * 2);        // 4.7 MB
  float*          h   = (float*)(ws + (size_t)MSZ * DSZ * 2
                                    + (size_t)DSZ * DSZ * 2);                 // 25.2 MB

  convert_f32_to_bf16<<<512, 256, 0, stream>>>(x,  xb,  MSZ * DSZ);
  convert_f32_to_bf16<<<512, 256, 0, stream>>>(w1, w1b, DSZ * DSZ);

  dim3 grid1(DSZ / TILE_N, MSZ / TILE_M);   // (24, 32)
  gemm1_gelu<<<grid1, 256, 0, stream>>>((const __bf16*)xb, (const __bf16*)w1b, b1, h);

  head_kernel<<<MSZ, 256, 0, stream>>>(h, ln_g, ln_b, w2, b2, affine, amask, out);
}
